// GNNLayer_37228776522275
// MI455X (gfx1250) — matrix operand order repaired and checked by reading, not compiled, not run on hardware
//
#include <hip/hip_runtime.h>
#include <hip/hip_fp16.h>

typedef _Float16 v16h __attribute__((ext_vector_type(16)));
typedef _Float16 v8h  __attribute__((ext_vector_type(8)));
typedef float    v8f  __attribute__((ext_vector_type(8)));

#define N_NODES 10000
#define N_EDGES 160000
#define DIM_IN  32
#define DIM_OUT 64
#define DIM_H   128
#define BOND    13
#define N_TILES (N_EDGES / 16)   // 10000

// ---------------------------------------------------------------------------
// Kernel 1: per-node prep.  x16 (f16 copy of x), base = x@root + bias -> d_out,
// xb = x@b2 (bias correction for messages).
// ---------------------------------------------------------------------------
__global__ __launch_bounds__(64) void prep_nodes(
    const float* __restrict__ x, const float* __restrict__ root,
    const float* __restrict__ bias, const float* __restrict__ b2,
    _Float16* __restrict__ x16, float* __restrict__ xb,
    float* __restrict__ out)
{
    __shared__ float sx[DIM_IN];
    int n = blockIdx.x;
    int t = threadIdx.x;
    if (t < DIM_IN) {
        float v = x[(size_t)n * DIM_IN + t];
        sx[t] = v;
        x16[(size_t)n * DIM_IN + t] = (_Float16)v;
    }
    __syncthreads();
    // t = output column o (0..63)
    float accR = bias[t];
    float accB = 0.0f;
    #pragma unroll
    for (int i = 0; i < DIM_IN; ++i) {
        float xv = sx[i];
        accR = fmaf(xv, root[i * DIM_OUT + t], accR);
        accB = fmaf(xv, b2[i * DIM_OUT + t], accB);
    }
    out[(size_t)n * DIM_OUT + t] = accR;   // pre-ReLU accumulator base
    xb [(size_t)n * DIM_OUT + t] = accB;
}

// ---------------------------------------------------------------------------
// Kernel 2: edge MLP  h = relu(edge_attr @ W1 + b1), stored f16 in the
// WMMA A-fragment element order (per 32-block permutation from ISA table:
// lane-group lo=(kk>>3)&1, element j=(kk&7)+((kk>>4)&1)*8).
// 2 edges per 256-thread block.
// ---------------------------------------------------------------------------
__global__ __launch_bounds__(256) void edge_mlp(
    const float* __restrict__ ea, const float* __restrict__ W1,
    const float* __restrict__ b1, _Float16* __restrict__ hpre)
{
    __shared__ float sea[2][BOND];
    int sub = threadIdx.x >> 7;          // 0/1: which edge in block
    int t   = threadIdx.x & 127;         // hidden unit k
    int e   = blockIdx.x * 2 + sub;
    if (t < BOND) sea[sub][t] = ea[(size_t)e * BOND + t];
    __syncthreads();
    float acc = b1[t];
    #pragma unroll
    for (int k = 0; k < BOND; ++k)
        acc = fmaf(sea[sub][k], W1[k * DIM_H + t], acc);
    acc = fmaxf(acc, 0.0f);
    int kc = t >> 5, kk = t & 31;
    int lo = (kk >> 3) & 1;
    int j  = (kk & 7) + ((kk >> 4) & 1) * 8;
    hpre[(size_t)e * DIM_H + kc * 32 + lo * 16 + j] = (_Float16)acc;
}

// ---------------------------------------------------------------------------
// Kernel 3: prepack W2 (f32 [128,2048]) into f16 B-fragments.
// Fragment f = (chunk c)*4 + ot, chunk c = i*4 + kc (kappa = i*128 + k).
// Dense-B lane layout: lanes 0-15 -> col n=lane, K=0..7,16..23 (elem j);
//                      lanes 16-31 -> col n=lane-16, K=8..15,24..31.
// 128 chunks * 4 otiles * 32 lanes * 16 halves = 262144 elements.
// ---------------------------------------------------------------------------
__global__ __launch_bounds__(256) void pack_w2(
    const float* __restrict__ W2, _Float16* __restrict__ bfrag)
{
    int idx  = blockIdx.x * 256 + threadIdx.x;     // 0..262143
    int j    = idx & 15;
    int lane = (idx >> 4) & 31;
    int ot   = (idx >> 9) & 3;
    int c    = idx >> 11;                          // 0..127
    int i    = c >> 2;
    int kc   = c & 3;
    int k    = kc * 32 + ((lane >= 16) ? 8 : 0) + (j & 7) + (j >> 3) * 16;
    int n    = ot * 16 + (lane & 15);
    bfrag[idx] = (_Float16)W2[(size_t)k * (DIM_IN * DIM_OUT) + i * DIM_OUT + n];
}

// ---------------------------------------------------------------------------
// Kernel 4: fused gather + (T @ B) GEMM + bias + scatter-add.
// 8 waves/block, one 16-edge tile per wave.  B (identical for all waves) is
// staged through LDS, double-buffered.  Per stage:
//   1. issue next stage's 4 coalesced uint4 global loads into regs (stg[])
//   2. issue next stage's x-scalar load (s_nxt) -- consumed NEXT stage
//   3. run this stage's 16 WMMAs off ds_load_b128 (ping-ponged bregs/abuf)
//   4. ds_store stg[] (its s_wait_loadcnt lands after all WMMAs), barrier
// So neither the B path nor the scalar path exposes global latency inside the
// WMMA stream; only short ds waits remain.
// ---------------------------------------------------------------------------
__global__ __launch_bounds__(256) void nnconv_main(
    const long long* __restrict__ ei,
    const _Float16* __restrict__ x16,
    const _Float16* __restrict__ hpre,
    const _Float16* __restrict__ bfrag,
    const float* __restrict__ xb,
    float* __restrict__ out)
{
    __shared__ uint4 sB[2][1024];        // 2 x 16 KB stage buffers

    int tid  = threadIdx.x;
    int wave = (blockIdx.x * blockDim.x + tid) >> 5;
    int lane = tid & 31;
    int row  = lane & 15;
    int lo   = lane >> 4;
    int e    = wave * 16 + row;          // grid is exact: wave < N_TILES always

    int src = (int)ei[e];
    const _Float16* x16row = x16 + (size_t)src * DIM_IN;

    // h fragments (A-side vectors), one per kc
    const v16h* hp = (const v16h*)(hpre + (size_t)e * DIM_H + lo * 16);
    v16h hf[4];
    hf[0] = hp[0]; hf[1] = hp[2]; hf[2] = hp[4]; hf[3] = hp[6];

    const uint4* gB = (const uint4*)bfrag;   // 32768 uint4 total (512 KB)

    v8f z = {0.f,0.f,0.f,0.f,0.f,0.f,0.f,0.f};
    v8f acc[4] = {z, z, z, z};
    v16h abuf[2];
    v16h bregs[2][4];

    // prologue: stage 0 into LDS, scalar for i=0 into register
    #pragma unroll
    for (int q = 0; q < 4; ++q)
        sB[0][q * 256 + tid] = gB[q * 256 + tid];
    _Float16 s_cur = x16row[0];
    __syncthreads();

    for (int st = 0; st < 32; ++st) {    // stage st == i value st
        int buf = st & 1;

        // (1) next stage's B into registers; (2) next stage's scalar
        uint4 stg[4];
        _Float16 s_nxt = s_cur;
        if (st < 31) {
            const uint4* gs = gB + (size_t)(st + 1) * 1024;
            #pragma unroll
            for (int q = 0; q < 4; ++q) stg[q] = gs[q * 256 + tid];
            s_nxt = x16row[st + 1];
        }
        if (st < 30)                     // keep HBM->L2 ahead on first pass
            __builtin_prefetch((const void*)(gB + (size_t)(st + 2) * 1024), 0, 1);

        // (3) compute: 4 chunks x 4 o-tiles
        const uint4* sbase = &sB[buf][(size_t)lane * 2];
        #pragma unroll
        for (int cc = 0; cc < 4; ++cc) { // chunk c = st*4+cc, kc == cc
            const int p = cc & 1;        // compile-time ping-pong parity
            abuf[p] = hf[cc] * s_cur;    // 8x v_pk_mul_f16
            #pragma unroll
            for (int ot = 0; ot < 4; ++ot)
                bregs[p][ot] = *(const v16h*)(sbase + (size_t)(cc * 4 + ot) * 64);
            #pragma unroll
            for (int ot = 0; ot < 4; ++ot)
                acc[ot] = __builtin_amdgcn_wmma_f32_16x16x32_f16(
                    false, abuf[p], false, bregs[p][ot], (short)0, acc[ot],
                    false, false);
        }

        // (4) commit next stage to LDS after this stage's WMMAs are in flight
        if (st < 31) {
            #pragma unroll
            for (int q = 0; q < 4; ++q)
                sB[buf ^ 1][q * 256 + tid] = stg[q];
        }
        s_cur = s_nxt;
        __syncthreads();
    }

    // Epilogue: C layout -> lane group lo owns rows M = r + 8*lo, col N = lane&15
    int n = lane & 15;
    int srcM[8], dstM[8];
    #pragma unroll
    for (int r = 0; r < 8; ++r) {
        int em = wave * 16 + lo * 8 + r;
        srcM[r] = (int)ei[em];
        dstM[r] = (int)ei[N_EDGES + em];
    }
    #pragma unroll
    for (int ot = 0; ot < 4; ++ot) {
        int o = ot * 16 + n;
        #pragma unroll
        for (int r = 0; r < 8; ++r) {
            float v = acc[ot][r] + xb[(size_t)srcM[r] * DIM_OUT + o];
            unsafeAtomicAdd(out + (size_t)dstM[r] * DIM_OUT + o, v);
        }
    }
}

// ---------------------------------------------------------------------------
// Kernel 5: final ReLU in place on d_out.
// ---------------------------------------------------------------------------
__global__ __launch_bounds__(256) void relu_out(float* __restrict__ out, int n)
{
    int i = blockIdx.x * 256 + threadIdx.x;
    if (i < n) out[i] = fmaxf(out[i], 0.0f);
}

// ---------------------------------------------------------------------------
extern "C" void kernel_launch(void* const* d_in, const int* in_sizes, int n_in,
                              void* d_out, int out_size, void* d_ws, size_t ws_size,
                              hipStream_t stream) {
    const float*     x    = (const float*)d_in[0];
    const long long* ei   = (const long long*)d_in[1];
    const float*     ea   = (const float*)d_in[2];
    const float*     W1   = (const float*)d_in[3];
    const float*     b1   = (const float*)d_in[4];
    const float*     W2   = (const float*)d_in[5];
    const float*     b2   = (const float*)d_in[6];
    const float*     root = (const float*)d_in[7];
    const float*     bias = (const float*)d_in[8];
    float* out = (float*)d_out;

    char* w = (char*)d_ws;
    _Float16* x16   = (_Float16*)(w);                       //   640 KB
    float*    xb    = (float*)   (w + 655360);              //  2560 KB
    _Float16* hpre  = (_Float16*)(w + 3215360);             // 40960 KB
    _Float16* bfrag = (_Float16*)(w + 44175360);            //   512 KB
    // total workspace: ~44.7 MB

    prep_nodes <<<N_NODES,        64, 0, stream>>>(x, root, bias, b2, x16, xb, out);
    edge_mlp   <<<N_EDGES / 2,   256, 0, stream>>>(ea, W1, b1, hpre);
    pack_w2    <<<262144 / 256,  256, 0, stream>>>(W2, bfrag);
    nnconv_main<<<N_TILES / 8,   256, 0, stream>>>(ei, x16, hpre, bfrag, xb, out);
    int tot = N_NODES * DIM_OUT;
    relu_out   <<<(tot + 255) / 256, 256, 0, stream>>>(out, tot);
}
